// GCN_17428977287424
// MI455X (gfx1250) — matrix operand order repaired and checked by reading, not compiled, not run on hardware
//
#include <hip/hip_runtime.h>
#include <hip/hip_bf16.h>

typedef __attribute__((ext_vector_type(16))) _Float16 v16h;
typedef __attribute__((ext_vector_type(8)))  float    v8f;

#define FDIM 64   // F_IN == HID == 64

// ---------------------------------------------------------------------------
// Zero a float buffer
// ---------------------------------------------------------------------------
__global__ void zero_f32(float* __restrict__ p, long n) {
    long t = (long)blockIdx.x * blockDim.x + threadIdx.x;
    if (t < n) p[t] = 0.0f;
}

// ---------------------------------------------------------------------------
// Degree / inverse-sqrt-degree (self-loop included as init=1)
// ---------------------------------------------------------------------------
__global__ void init_deg(float* __restrict__ deg, int n) {
    int t = blockIdx.x * blockDim.x + threadIdx.x;
    if (t < n) deg[t] = 1.0f;               // self-loop
}
__global__ void count_deg(const int* __restrict__ dst, float* __restrict__ deg, int e) {
    int t = blockIdx.x * blockDim.x + threadIdx.x;
    if (t < e) atomicAdd(&deg[dst[t]], 1.0f);
}
__global__ void finish_dinv(float* __restrict__ deg, int n) {
    int t = blockIdx.x * blockDim.x + threadIdx.x;
    if (t < n) { float d = deg[t]; deg[t] = (d > 0.0f) ? rsqrtf(d) : 0.0f; }
}

// ---------------------------------------------------------------------------
// GEMM: O[N,64] = H[N,64] @ W[64,64]  via v_wmma_f32_16x16x32_f16
// One wave -> 16 rows x 64 cols; 8 waves / block.
// W is staged TRANSPOSED in LDS (Ws[n][k]) so each lane's B fragment is a
// contiguous 32-byte run -> 2x ds_load_b128 instead of 16x ds_load_u16.
// ---------------------------------------------------------------------------
__global__ __launch_bounds__(256) void gemm_n64(const float* __restrict__ H,
                                                const float* __restrict__ W,
                                                float* __restrict__ O, int N) {
    __shared__ alignas(32) _Float16 Ws[FDIM * FDIM];   // [n][k] (transposed)
    for (int i = threadIdx.x; i < FDIM * FDIM; i += 256) {
        const int k = i >> 6;
        const int n = i & 63;
        Ws[n * FDIM + k] = (_Float16)W[i];             // W is [k][n] row-major
    }
    __syncthreads();

    const int wave = threadIdx.x >> 5;
    const int lane = threadIdx.x & 31;
    const int m0 = (blockIdx.x * 8 + wave) * 16;
    if (m0 >= N) return;                    // wave-uniform: EXEC stays all-ones

    const int half = lane >> 4;             // 0: lanes 0-15, 1: lanes 16-31
    const int l16  = lane & 15;
    int rowA = m0 + l16;
    if (rowA >= N) rowA = N - 1;            // clamp loads, keep EXEC full
    const float* __restrict__ hrow = H + (size_t)rowA * FDIM;

    v8f acc[4] = {};
    #pragma unroll
    for (int kk = 0; kk < 2; ++kk) {        // K = 64 in two 32-steps
        const int koff = kk * 32;
        // A fragment (contiguous K, matching B): lane<16 -> K{0..15}, lane>=16 -> K{16..31}
        v16h a;
        const int kbA = koff + half * 16;
        #pragma unroll
        for (int i = 0; i < 8; ++i) {
            a[i]     = (_Float16)hrow[kbA + i];
            a[i + 8] = (_Float16)hrow[kbA + 8 + i];
        }
        const int kbB = koff + half * 16;   // lane's contiguous K range for B
        #pragma unroll
        for (int nt = 0; nt < 4; ++nt) {    // four 16-col tiles
            const int colB = nt * 16 + l16;
            // B fragment (32x16): lane holds column colB; 16 contiguous f16 in LDS
            const v16h b = *(const v16h*)&Ws[colB * FDIM + kbB];
            acc[nt] = __builtin_amdgcn_wmma_f32_16x16x32_f16(
                false, a, false, b, (short)0, acc[nt], false, false);
        }
    }
    // C/D layout: VGPR r -> M = m0 + r + 8*half, N = n0 + l16
    #pragma unroll
    for (int nt = 0; nt < 4; ++nt) {
        const int n0 = nt * 16;
        #pragma unroll
        for (int r = 0; r < 8; ++r) {
            const int row = m0 + r + half * 8;
            if (row < N) O[(size_t)row * FDIM + n0 + l16] = acc[nt][r];
        }
    }
}

// ---------------------------------------------------------------------------
// Edge scatter: agg[dst] += T[src] * dinv[src]*dinv[dst]
// 16 lanes per edge, float4 per lane.
// ---------------------------------------------------------------------------
__global__ void scatter_edges(const float* __restrict__ T,
                              const int* __restrict__ src,
                              const int* __restrict__ dst,
                              const float* __restrict__ dinv,
                              float* __restrict__ agg, int E) {
    long t = (long)blockIdx.x * blockDim.x + threadIdx.x;
    int e = (int)(t >> 4);
    int c = ((int)t & 15) * 4;
    if (e >= E) return;
    int s = src[e], d = dst[e];
    float nrm = dinv[s] * dinv[d];
    const float4 v = *(const float4*)(T + (size_t)s * FDIM + c);
    float* ap = agg + (size_t)d * FDIM + c;
    atomicAdd(ap + 0, v.x * nrm);
    atomicAdd(ap + 1, v.y * nrm);
    atomicAdd(ap + 2, v.z * nrm);
    atomicAdd(ap + 3, v.w * nrm);
}

// ---------------------------------------------------------------------------
// Epilogue: H = relu(agg + selfloop(T*dinv^2) + bias), in place over agg.
// ---------------------------------------------------------------------------
__global__ void finish_layer(float* __restrict__ agg,
                             const float* __restrict__ T,
                             const float* __restrict__ dinv,
                             const float* __restrict__ b, long n64) {
    long t = (long)blockIdx.x * blockDim.x + threadIdx.x;
    if (t >= n64) return;
    int node = (int)(t >> 6);
    int f    = (int)t & 63;
    float di = dinv[node];
    float v = agg[t] + T[t] * di * di + b[f];
    agg[t] = fmaxf(v, 0.0f);
}

// ---------------------------------------------------------------------------
// Pooling scatter: sums[batch[n]] += H[n]; cnt[batch[n]] += 1
// ---------------------------------------------------------------------------
__global__ void pool_scatter(const float* __restrict__ H,
                             const int* __restrict__ batch,
                             float* __restrict__ sums,
                             float* __restrict__ cnt, int N) {
    long t = (long)blockIdx.x * blockDim.x + threadIdx.x;
    if (t >= (long)N * FDIM) return;
    int node = (int)(t >> 6);
    int f    = (int)t & 63;
    int g = batch[node];
    atomicAdd(&sums[(size_t)g * FDIM + f], H[t]);
    if (f == 0) atomicAdd(&cnt[g], 1.0f);
}

// ---------------------------------------------------------------------------
// Heads: pooled = sums/max(cnt,1); out[g] = pooled.Wmem + bmem, out[G+g] = pooled.Wtime + btime
// ---------------------------------------------------------------------------
__global__ void heads(const float* __restrict__ sums, const float* __restrict__ cnt,
                      const float* __restrict__ Wmem, const float* __restrict__ bmem,
                      const float* __restrict__ Wtime, const float* __restrict__ btime,
                      float* __restrict__ out, int G) {
    int g = blockIdx.x * blockDim.x + threadIdx.x;
    if (g >= G) return;
    float c = fmaxf(cnt[g], 1.0f);
    float inv = 1.0f / c;
    float m = 0.0f, ti = 0.0f;
    #pragma unroll
    for (int f = 0; f < FDIM; ++f) {
        float p = sums[(size_t)g * FDIM + f] * inv;
        m  += p * Wmem[f];
        ti += p * Wtime[f];
    }
    out[g]     = m  + bmem[0];
    out[G + g] = ti + btime[0];
}

// ---------------------------------------------------------------------------
extern "C" void kernel_launch(void* const* d_in, const int* in_sizes, int n_in,
                              void* d_out, int out_size, void* d_ws, size_t ws_size,
                              hipStream_t stream) {
    const float* x     = (const float*)d_in[0];
    const int*   ei    = (const int*)d_in[1];
    const int*   batch = (const int*)d_in[2];
    const float* W1    = (const float*)d_in[3];
    const float* b1    = (const float*)d_in[4];
    const float* W2    = (const float*)d_in[5];
    const float* b2    = (const float*)d_in[6];
    const float* Wmem  = (const float*)d_in[7];
    const float* bmem  = (const float*)d_in[8];
    const float* Wtime = (const float*)d_in[9];
    const float* btime = (const float*)d_in[10];

    const int N = in_sizes[0] / FDIM;        // 50000
    const int E = in_sizes[1] / 2;           // 800000
    const int G = out_size / 2;              // 512
    const int* src = ei;
    const int* dst = ei + E;

    // workspace layout (floats)
    float* ws = (float*)d_ws;
    const long nf = (long)N * FDIM;
    float* Tbuf = ws;                        // N*64  transformed
    float* Abuf = Tbuf + nf;                 // N*64  aggregate / activation
    float* dinv = Abuf + nf;                 // N
    float* sums = dinv + N;                  // G*64
    float* cnt  = sums + (long)G * FDIM;     // G

    const int TB = 256;
    auto blks = [](long n, int tb) { return (int)((n + tb - 1) / tb); };

    // degree -> dinv
    init_deg<<<blks(N, TB), TB, 0, stream>>>(dinv, N);
    count_deg<<<blks(E, TB), TB, 0, stream>>>(dst, dinv, E);
    finish_dinv<<<blks(N, TB), TB, 0, stream>>>(dinv, N);

    const int gemmBlocks = (N + 16 * 8 - 1) / (16 * 8);
    const long edgeThreads = (long)E * 16;

    // ---- layer 1: x -> Abuf
    gemm_n64<<<gemmBlocks, 256, 0, stream>>>(x, W1, Tbuf, N);
    zero_f32<<<blks(nf, TB), TB, 0, stream>>>(Abuf, nf);
    scatter_edges<<<blks(edgeThreads, TB), TB, 0, stream>>>(Tbuf, src, dst, dinv, Abuf, E);
    finish_layer<<<blks(nf, TB), TB, 0, stream>>>(Abuf, Tbuf, dinv, b1, nf);

    // ---- layers 2 & 3: Abuf -> Abuf (A consumed into T before A is re-zeroed)
    for (int layer = 1; layer < 3; ++layer) {
        gemm_n64<<<gemmBlocks, 256, 0, stream>>>(Abuf, W2, Tbuf, N);
        zero_f32<<<blks(nf, TB), TB, 0, stream>>>(Abuf, nf);
        scatter_edges<<<blks(edgeThreads, TB), TB, 0, stream>>>(Tbuf, src, dst, dinv, Abuf, E);
        finish_layer<<<blks(nf, TB), TB, 0, stream>>>(Abuf, Tbuf, dinv, b2, nf);
    }

    // ---- pooling + heads
    zero_f32<<<blks((long)G * FDIM + G, TB), TB, 0, stream>>>(sums, (long)G * FDIM + G);
    pool_scatter<<<blks((long)N * FDIM, TB), TB, 0, stream>>>(Abuf, batch, sums, cnt, N);
    heads<<<blks(G, TB), TB, 0, stream>>>(sums, cnt, Wmem, bmem, Wtime, btime,
                                          (float*)d_out, G);
}
